// MultiHeadAttention_84499186582008
// MI455X (gfx1250) — matrix operand order repaired and checked by reading, not compiled, not run on hardware
//
#include <hip/hip_runtime.h>
#include <hip/hip_bf16.h>

// ---------------------------------------------------------------------------
// MHA for MI455X (gfx1250): bf16 WMMA pipeline.
//   phase 0: f32 -> bf16 conversion of x, qkv_w, o_w          (bandwidth)
//   phase 1: QKV GEMM (WMMA) -> Q*(1/sqrt(D)), K, V^T scatter (compute)
//   phase 2: flash-attention, block = one (b,h) x 128 q rows  (compute)
//   phase 3: output projection GEMM (WMMA) + bias -> f32 out  (compute)
// All LDS staging uses GLOBAL_LOAD_ASYNC_TO_LDS_B128 with double-buffered
// tiles: DMA of tile i+1 overlaps WMMA compute of tile i (ASYNCcnt pipelined).
// ---------------------------------------------------------------------------

typedef unsigned short u16;
typedef unsigned int   u32;
typedef unsigned long long u64;
typedef __attribute__((ext_vector_type(16))) __bf16 v16bf;
typedef __attribute__((ext_vector_type(8)))  float  v8f;
typedef __attribute__((ext_vector_type(16))) u16    v16u;
typedef __attribute__((ext_vector_type(4)))  u32    v4u;

#define EMBED 1024
#define HEADS 16
#define HD    64
#define SEQ   2048
#define BATCH 4
#define MROWS (BATCH*SEQ)   // 8192

__device__ __forceinline__ u16 f2bf(float f) {
  u32 u = __builtin_bit_cast(u32, f);
  u32 r = u + 0x7FFFu + ((u >> 16) & 1u);   // round-to-nearest-even
  return (u16)(r >> 16);
}
__device__ __forceinline__ v16bf asbf(v16u v) { return __builtin_bit_cast(v16bf, v); }

__device__ __forceinline__ v8f wmma_bf16(v16u a, v16u b, v8f c) {
  return __builtin_amdgcn_wmma_f32_16x16x32_bf16(
      false, asbf(a), false, asbf(b), (short)0, c, false, false);
}

// CDNA5 async DMA: global -> LDS, 16B per lane, tracked by ASYNCcnt.
// LDS aperture addressing (ISA 10.2): generic shared addr low 32 bits == LDS
// byte offset, which is exactly what VDST of the async-load wants.
__device__ __forceinline__ void async_g2l_b128(const void* gptr, void* lptr) {
  asm volatile("global_load_async_to_lds_b128 %0, %1, off"
               :: "v"((u32)(uintptr_t)lptr),
                  "v"((u64)(uintptr_t)gptr)
               : "memory");
}
__device__ __forceinline__ void wait_async0() {
  asm volatile("s_wait_asynccnt 0" ::: "memory");
}

// max-reduction across a 16-lane row, pure VALU via DPP butterfly
__device__ __forceinline__ float dpp_rowmax16(float x) {
  int xi, yi;
  xi = __builtin_bit_cast(int, x);
  yi = __builtin_amdgcn_mov_dpp(xi, 0xB1, 0xF, 0xF, true);   // quad_perm(1,0,3,2)
  x = fmaxf(x, __builtin_bit_cast(float, yi));
  xi = __builtin_bit_cast(int, x);
  yi = __builtin_amdgcn_mov_dpp(xi, 0x4E, 0xF, 0xF, true);   // quad_perm(2,3,0,1)
  x = fmaxf(x, __builtin_bit_cast(float, yi));
  xi = __builtin_bit_cast(int, x);
  yi = __builtin_amdgcn_mov_dpp(xi, 0x141, 0xF, 0xF, true);  // row_half_mirror
  x = fmaxf(x, __builtin_bit_cast(float, yi));
  xi = __builtin_bit_cast(int, x);
  yi = __builtin_amdgcn_mov_dpp(xi, 0x140, 0xF, 0xF, true);  // row_mirror
  x = fmaxf(x, __builtin_bit_cast(float, yi));
  return x;
}

// A-fragment 16x32 bf16 from row-major [16][stride]:
// lanes 0-15: halves = K{0..7,16..23}; lanes 16-31: K{8..15,24..31} (ISA 7.12.2)
__device__ __forceinline__ v16u frag_a(const u16* base, int stride, int lane) {
  int r  = lane & 15;
  int kb = (lane & 16) ? 8 : 0;
  const u16* p = base + r * stride + kb;
  v16u o;
  ((v4u*)&o)[0] = *(const v4u*)(p);
  ((v4u*)&o)[1] = *(const v4u*)(p + 16);
  return o;
}
// B-fragment 32x16 bf16 from B^T row-major [16][stride]:
// lanes 0-15: K{0..7,16..23}; lanes 16-31: K{8..15,24..31} (same as A)
__device__ __forceinline__ v16u frag_b(const u16* base, int stride, int lane) {
  int n  = lane & 15;
  int kb = (lane & 16) ? 8 : 0;
  const u16* p = base + n * stride + kb;
  v16u o;
  ((v4u*)&o)[0] = *(const v4u*)(p);
  ((v4u*)&o)[1] = *(const v4u*)(p + 16);
  return o;
}

// ------------------------------- phase 0 -----------------------------------
__global__ __launch_bounds__(256) void k_cvt(const float* __restrict__ s,
                                             u16* __restrict__ d, int n) {
  int i = (blockIdx.x * blockDim.x + threadIdx.x) * 4;
  if (i < n) {
    float4 f = *(const float4*)(s + i);
    ushort4 o;
    o.x = f2bf(f.x); o.y = f2bf(f.y); o.z = f2bf(f.z); o.w = f2bf(f.w);
    *(ushort4*)(d + i) = o;
  }
}

// --------------------------- phases 1 & 3: GEMM ----------------------------
// C[m,n] = sum_k A[m,k] * Bw[n,k]  (Bw stored row-major [N][K] == B^T)
// MODE 0: QKV epilogue (bias + scatter to Q/K/V^T, Q pre-scaled by 1/8)
// MODE 1: plain f32 epilogue (bias) -> Cout
template <int MODE>
__global__ __launch_bounds__(256) void k_gemm(
    const u16* __restrict__ A, const u16* __restrict__ Bw,
    const float* __restrict__ bias, float* __restrict__ Cout,
    u16* __restrict__ Qo, u16* __restrict__ Ko, u16* __restrict__ Vo,
    int Kdim, int Ndim) {
  __shared__ u16 As[2][128 * 40];   // double-buffered 128x32 tile (pad 40)
  __shared__ u16 Bs[2][64 * 40];    // double-buffered  64x32 tile

  const int t    = threadIdx.x;
  const int lane = t & 31, wave = t >> 5;
  const int m0 = blockIdx.y * 128, n0 = blockIdx.x * 64;
  const int wm = (wave >> 1) * 32, wn = (wave & 1) * 32;

  // per-thread staging coordinates (A: 2 chunks, B: 1 chunk, 16B each)
  const int ar0 = t >> 2,           ao0 = (t & 3) << 3;
  const int ar1 = (t + 256) >> 2,   ao1 = ao0;
  const int br  = t >> 2,           bo  = (t & 3) << 3;

  v8f acc[2][2] = {};

  // prologue: DMA tile 0
  async_g2l_b128(A + (size_t)(m0 + ar0) * Kdim + ao0, &As[0][ar0 * 40 + ao0]);
  async_g2l_b128(A + (size_t)(m0 + ar1) * Kdim + ao1, &As[0][ar1 * 40 + ao1]);
  async_g2l_b128(Bw + (size_t)(n0 + br) * Kdim + bo, &Bs[0][br * 40 + bo]);
  wait_async0();
  __syncthreads();

  int buf = 0;
  for (int k0 = 0; k0 < Kdim; k0 += 32) {
    // issue DMA for next tile into the inactive buffer (overlaps compute)
    if (k0 + 32 < Kdim) {
      int kn = k0 + 32;
      async_g2l_b128(A + (size_t)(m0 + ar0) * Kdim + kn + ao0,
                     &As[buf ^ 1][ar0 * 40 + ao0]);
      async_g2l_b128(A + (size_t)(m0 + ar1) * Kdim + kn + ao1,
                     &As[buf ^ 1][ar1 * 40 + ao1]);
      async_g2l_b128(Bw + (size_t)(n0 + br) * Kdim + kn + bo,
                     &Bs[buf ^ 1][br * 40 + bo]);
    }

    v16u a0 = frag_a(&As[buf][wm * 40], 40, lane);
    v16u a1 = frag_a(&As[buf][(wm + 16) * 40], 40, lane);
    v16u b0 = frag_b(&Bs[buf][wn * 40], 40, lane);
    v16u b1 = frag_b(&Bs[buf][(wn + 16) * 40], 40, lane);
    acc[0][0] = wmma_bf16(a0, b0, acc[0][0]);
    acc[0][1] = wmma_bf16(a0, b1, acc[0][1]);
    acc[1][0] = wmma_bf16(a1, b0, acc[1][0]);
    acc[1][1] = wmma_bf16(a1, b1, acc[1][1]);

    wait_async0();      // next tile landed
    __syncthreads();    // everyone done reading current tile
    buf ^= 1;
  }

  const int hi = (lane & 16) ? 8 : 0;
#pragma unroll
  for (int i = 0; i < 2; ++i)
#pragma unroll
    for (int j = 0; j < 2; ++j)
#pragma unroll
      for (int g = 0; g < 8; ++g) {
        int row = m0 + wm + i * 16 + g + hi;
        int col = n0 + wn + j * 16 + (lane & 15);
        float v = acc[i][j][g] + bias[col];
        if (MODE == 0) {
          int h = col / (3 * HD);
          int rr = col - h * (3 * HD);
          int sel = rr >> 6, d = rr & 63;
          int b = row >> 11, nq = row & (SEQ - 1);
          int bh = b * HEADS + h;
          if (sel == 0)                                   // Q, fold 1/sqrt(D)
            Qo[((size_t)bh * SEQ + nq) * HD + d] = f2bf(v * 0.125f);
          else if (sel == 1)                              // K [bh,n,d]
            Ko[((size_t)bh * SEQ + nq) * HD + d] = f2bf(v);
          else                                            // V^T [bh,d,n]
            Vo[((size_t)bh * HD + d) * SEQ + nq] = f2bf(v);
        } else {
          Cout[(size_t)row * Ndim + col] = v;
        }
      }
}

// ------------------------ phase 2: flash attention -------------------------
// block = one (b,h) slice x 128 q rows (8 waves x 16); kv step 32.
// Double-buffered K/V chunks via async DMA, shared by all 8 waves.
__global__ __launch_bounds__(256) void k_attn(const u16* __restrict__ Q,
                                              const u16* __restrict__ K,
                                              const u16* __restrict__ Vt,
                                              u16* __restrict__ AO) {
  __shared__ u16 Ks[2][32 * 72];   // K chunk  [n=32][d=64] (+pad)
  __shared__ u16 Vs[2][64 * 40];   // V^T chunk [d=64][n=32] (+pad)
  __shared__ u16 Pb[8][16 * 40];   // per-wave P-tile transpose scratch

  const int t = threadIdx.x, lane = t & 31, wave = t >> 5;
  const int bh = blockIdx.x >> 4;        // 0..63
  const int qb = blockIdx.x & 15;        // q block of 128 rows
  const int q0 = qb * 128 + wave * 16;

  const u16* Qp = Q + ((size_t)bh * SEQ + q0) * HD;
  const u16* Kp = K + (size_t)bh * SEQ * HD;
  const u16* Vp = Vt + (size_t)bh * HD * SEQ;
  u16* P = &Pb[wave][0];

  const int r = lane & 15;
  const int hilane = (lane & 16) ? 1 : 0;
  const int ka  = hilane ? 8 : 0;    // A-frag interleaved base
  const int kbb = hilane ? 8 : 0;    // B-frag interleaved base

  // per-thread staging coordinates (K: 1 chunk, V: 1 chunk, 16B each)
  const int kr = t >> 3, ko = (t & 7) << 3;
  const int vr = t >> 2, vo_ = (t & 3) << 3;

  // Q A-fragments (d 0..31 and 32..63), already scaled by 1/sqrt(D)
  v16u aq0, aq1;
  {
    const u16* p = Qp + r * HD;
    ((v4u*)&aq0)[0] = *(const v4u*)(p + ka);
    ((v4u*)&aq0)[1] = *(const v4u*)(p + ka + 16);
    ((v4u*)&aq1)[0] = *(const v4u*)(p + 32 + ka);
    ((v4u*)&aq1)[1] = *(const v4u*)(p + 32 + ka + 16);
  }

  // all-1.0 bf16 B-fragment for the row-sum WMMA trick
  v16u ones;
#pragma unroll
  for (int i = 0; i < 16; ++i) ones[i] = 0x3F80;

  float mrow[8], lrow[8], cf[8];
#pragma unroll
  for (int g = 0; g < 8; ++g) { mrow[g] = -3.0e38f; lrow[g] = 0.f; }
  v8f vo[4] = {};

  // prologue: DMA chunk 0
  async_g2l_b128(Kp + (size_t)kr * HD + ko, &Ks[0][kr * 72 + ko]);
  async_g2l_b128(Vp + (size_t)vr * SEQ + vo_, &Vs[0][vr * 40 + vo_]);
  wait_async0();
  __syncthreads();

  int buf = 0;
  for (int kv = 0; kv < SEQ; kv += 32) {
    // issue DMA for next chunk into the inactive buffer (overlaps compute)
    if (kv + 32 < SEQ) {
      async_g2l_b128(Kp + (size_t)(kv + 32 + kr) * HD + ko,
                     &Ks[buf ^ 1][kr * 72 + ko]);
      async_g2l_b128(Vp + (size_t)vr * SEQ + kv + 32 + vo_,
                     &Vs[buf ^ 1][vr * 40 + vo_]);
    }

    // S = Q K^T : two 16x16 score tiles (cols kv..+15, kv+16..+31)
    v8f s0 = {}, s1 = {};
    {
      const u16* k0p = &Ks[buf][r * 72];
      const u16* k1p = &Ks[buf][(16 + r) * 72];
      v16u bk;
      ((v4u*)&bk)[0] = *(const v4u*)(k0p + kbb);
      ((v4u*)&bk)[1] = *(const v4u*)(k0p + kbb + 16);
      s0 = wmma_bf16(aq0, bk, s0);
      ((v4u*)&bk)[0] = *(const v4u*)(k0p + 32 + kbb);
      ((v4u*)&bk)[1] = *(const v4u*)(k0p + 32 + kbb + 16);
      s0 = wmma_bf16(aq1, bk, s0);
      ((v4u*)&bk)[0] = *(const v4u*)(k1p + kbb);
      ((v4u*)&bk)[1] = *(const v4u*)(k1p + kbb + 16);
      s1 = wmma_bf16(aq0, bk, s1);
      ((v4u*)&bk)[0] = *(const v4u*)(k1p + 32 + kbb);
      ((v4u*)&bk)[1] = *(const v4u*)(k1p + 32 + kbb + 16);
      s1 = wmma_bf16(aq1, bk, s1);
    }

    // online softmax: rows in VGPR index, cols in 16-lane halves.
    // row max via DPP butterfly (no LDS); row sum via WMMA below.
#pragma unroll
    for (int g = 0; g < 8; ++g) {
      float mx = dpp_rowmax16(fmaxf(s0[g], s1[g]));
      float mn = fmaxf(mrow[g], mx);
      cf[g] = __expf(mrow[g] - mn);
      mrow[g] = mn;
      float p0 = __expf(s0[g] - mn);
      float p1 = __expf(s1[g] - mn);
      int pr = g + hilane * 8;
      P[pr * 40 + r]      = f2bf(p0);   // transpose C-layout -> row-major
      P[pr * 40 + 16 + r] = f2bf(p1);
    }

    // P A-fragment from per-wave LDS (same-wave DS ops are in-order)
    v16u ap;
    {
      const u16* pp = P + r * 40 + ka;
      ((v4u*)&ap)[0] = *(const v4u*)pp;
      ((v4u*)&ap)[1] = *(const v4u*)(pp + 16);
    }

    // row sums of P via WMMA against all-ones: lands in [g] layout directly
    v8f zero = {};
    v8f sums = wmma_bf16(ap, ones, zero);
#pragma unroll
    for (int g = 0; g < 8; ++g) lrow[g] = lrow[g] * cf[g] + sums[g];

#pragma unroll
    for (int tt = 0; tt < 4; ++tt)
#pragma unroll
      for (int g = 0; g < 8; ++g) vo[tt][g] *= cf[g];

    // O += P V : V^T rows in LDS give contiguous B-fragments
#pragma unroll
    for (int tt = 0; tt < 4; ++tt) {
      const u16* vp = &Vs[buf][(tt * 16 + r) * 40 + kbb];
      v16u bv;
      ((v4u*)&bv)[0] = *(const v4u*)(vp);
      ((v4u*)&bv)[1] = *(const v4u*)(vp + 16);
      vo[tt] = wmma_bf16(ap, bv, vo[tt]);
    }

    wait_async0();      // next chunk landed
    __syncthreads();    // everyone done reading current chunk
    buf ^= 1;
  }

  // epilogue: normalize and scatter to [b, n, h*64+d] bf16
  const int b = bh >> 4, h = bh & 15;
#pragma unroll
  for (int tt = 0; tt < 4; ++tt)
#pragma unroll
    for (int g = 0; g < 8; ++g) {
      int row = q0 + g + hilane * 8;
      float val = vo[tt][g] / lrow[g];
      int col = h * HD + tt * 16 + r;
      AO[((size_t)b * SEQ + row) * EMBED + col] = f2bf(val);
    }
}

// ---------------------------------------------------------------------------
extern "C" void kernel_launch(void* const* d_in, const int* in_sizes, int n_in,
                              void* d_out, int out_size, void* d_ws, size_t ws_size,
                              hipStream_t stream) {
  const float* x     = (const float*)d_in[0];   // [4,2048,1024]
  const float* qkv_w = (const float*)d_in[1];   // [3072,1024]
  const float* qkv_b = (const float*)d_in[2];   // [3072]
  const float* o_w   = (const float*)d_in[3];   // [1024,1024]
  const float* o_b   = (const float*)d_in[4];   // [1024]
  float* out = (float*)d_out;                   // [4,2048,1024] f32

  // workspace layout (bytes): needs 88 MiB
  char* ws = (char*)d_ws;
  u16* xb   = (u16*)(ws);                               // x bf16      16 MiB
  u16* wqkv = (u16*)(ws + (size_t)16 * 1024 * 1024);    // qkv_w bf16   6 MiB
  u16* wo   = (u16*)(ws + (size_t)22 * 1024 * 1024);    // o_w bf16     2 MiB
  u16* Qb   = (u16*)(ws + (size_t)24 * 1024 * 1024);    // Q [bh,n,d]  16 MiB
  u16* Kb   = (u16*)(ws + (size_t)40 * 1024 * 1024);    // K [bh,n,d]  16 MiB
  u16* Vb   = (u16*)(ws + (size_t)56 * 1024 * 1024);    // V^T [bh,d,n]16 MiB
  u16* AO   = (u16*)(ws + (size_t)72 * 1024 * 1024);    // attn out    16 MiB

  // phase 0: conversions
  k_cvt<<<(8388608 / 4 + 255) / 256, 256, 0, stream>>>(x, xb, 8388608);
  k_cvt<<<(3145728 / 4 + 255) / 256, 256, 0, stream>>>(qkv_w, wqkv, 3145728);
  k_cvt<<<(1048576 / 4 + 255) / 256, 256, 0, stream>>>(o_w, wo, 1048576);

  // phase 1: QKV GEMM  (M=8192, N=3072, K=1024)
  dim3 g1(3072 / 64, MROWS / 128);
  k_gemm<0><<<g1, 256, 0, stream>>>(xb, wqkv, qkv_b, nullptr, Qb, Kb, Vb,
                                    1024, 3072);

  // phase 2: attention  (64 bh x 16 q-blocks of 128 rows)
  k_attn<<<64 * 16, 256, 0, stream>>>(Qb, Kb, Vb, AO);

  // phase 3: output projection  (M=8192, N=1024, K=1024) + bias -> f32
  dim3 g2(1024 / 64, MROWS / 128);
  k_gemm<1><<<g2, 256, 0, stream>>>(AO, wo, o_b, out, nullptr, nullptr, nullptr,
                                    1024, 1024);
}
